// VarianceKPNet_71708773974931
// MI455X (gfx1250) — matrix operand order repaired and checked by reading, not compiled, not run on hardware
//
#include <hip/hip_runtime.h>
#include <hip/hip_bf16.h>

typedef __attribute__((ext_vector_type(16))) _Float16     v16h;
typedef __attribute__((ext_vector_type(8)))  float        v8f;
typedef __attribute__((ext_vector_type(4)))  unsigned int u32x4;
typedef __attribute__((ext_vector_type(8)))  int          i32x8;
typedef __attribute__((ext_vector_type(4)))  int          i32x4;

#define H2 512
#define W2 512
#define NPT 4096

#ifndef USE_TDM
#define USE_TDM 1
#endif
#if __has_include(<hip/amd_detail/amd_gfx1250_TDM.h>)
#define TDM_6ARG 1
#endif

// ---------------- Tensor Data Mover: contiguous 2-byte-element block -> LDS ----------------
__device__ __forceinline__ void tdm_load_to_lds(const void* gsrc, unsigned ldsOff, unsigned nElem2B){
  unsigned long long ga = (unsigned long long)(size_t)gsrc;
  u32x4 g0 = {0,0,0,0};
  g0.x = 1u;                                            // count=1, user descriptor
  g0.y = ldsOff;                                        // lds_addr (bytes)
  g0.z = (unsigned)ga;                                  // global_addr[31:0]
  g0.w = (unsigned)((ga>>32)&0x01ffffffu) | (2u<<30);   // global_addr[56:32], type=2 ("image")
  i32x8 g1 = {0,0,0,0,0,0,0,0};
  g1[0] = (int)(1u<<16);                                // workgroup_mask=0, data_size=1 (2B)
  g1[1] = (int)((nElem2B & 0xffffu) << 16);             // tensor_dim0[15:0]
  g1[2] = (int)((nElem2B >> 16) & 0xffffu)              // tensor_dim0[31:16]
        | (int)(1u<<16);                                // tensor_dim1=1
  g1[3] = (int)((nElem2B & 0xffffu) << 16);             // tile_dim0
  g1[4] = 1;                                            // tile_dim1=1, tile_dim2=0
  g1[5] = (int)nElem2B;                                 // tensor_dim0_stride[31:0]
  g1[6] = 0; g1[7] = 0;
  i32x4 gz = {0,0,0,0};
#if defined(TDM_6ARG)
  i32x8 gz8 = {0,0,0,0,0,0,0,0};
  __builtin_amdgcn_tensor_load_to_lds(g0, g1, gz, gz, gz8, 0);
#else
  __builtin_amdgcn_tensor_load_to_lds(g0, g1, gz, gz, 0);
#endif
}

__device__ __forceinline__ void stage_weights(_Float16* wlds, const _Float16* gsrc, unsigned nElem2B){
#if USE_TDM
  if((threadIdx.x >> 5) == 0){
    tdm_load_to_lds(gsrc, (unsigned)(size_t)wlds, nElem2B);
    __builtin_amdgcn_s_wait_tensorcnt(0);
  }
  __syncthreads();
#else
  for(unsigned i = threadIdx.x; i < nElem2B/8; i += blockDim.x)
    ((u32x4*)wlds)[i] = ((const u32x4*)gsrc)[i];
  __syncthreads();
#endif
}

// ---------------- wave32 row reductions ----------------
__device__ inline float wred_max16(float v){
#pragma unroll
  for(int m=1;m<16;m<<=1) v = fmaxf(v, __shfl_xor(v, m, 32));
  return v;
}
__device__ inline float wred_sum16(float v){
#pragma unroll
  for(int m=1;m<16;m<<=1) v += __shfl_xor(v, m, 32);
  return v;
}
__device__ inline float wred_sum32(float v){
#pragma unroll
  for(int m=1;m<32;m<<=1) v += __shfl_xor(v, m, 32);
  return v;
}

// A-operand loader: two 16B vector loads -> f16 A layout
__device__ __forceinline__ v16h load_a_frag(const _Float16* base, bool inb){
  union { u32x4 u[2]; v16h v; } au;
  u32x4 z = {0,0,0,0};
  if(inb){
    const u32x4* p = (const u32x4*)base;
    au.u[0] = p[0];
    au.u[1] = p[1];
  } else { au.u[0] = z; au.u[1] = z; }
  return au.v;
}

// ---------------- weight repack: f32 OIHW/[N,K] -> f16 WMMA B-operand layout ----------------
template<int CIN, int COUT, int TAPS>
__global__ void k_repack(const float* __restrict__ w, _Float16* __restrict__ wb){
  const int NC = TAPS*CIN/32;
  int idx = blockIdx.x*blockDim.x + threadIdx.x;
  if(idx >= (COUT/16)*NC*512) return;
  int e     = idx & 15;
  int lane  = (idx>>4) & 31;
  int chunk = (idx>>9) % NC;
  int ct    = idx/(NC*512);
  int tap   = chunk/(CIN/32);
  int c0    = (chunk%(CIN/32))*32;
  int n     = ct*16 + (lane&15);
  int cin   = c0 + ((lane&16)?16:0) + e;
  float val = (TAPS==9) ? w[((size_t)n*CIN + cin)*9 + tap] : w[(size_t)n*CIN + cin];
  wb[idx] = (_Float16)val;
}

// ---------------- bilinear 2x upsample, NCHW f32 -> HWC f16 ----------------
__global__ void k_upsample(const float* __restrict__ x, _Float16* __restrict__ out){
  int idx = blockIdx.x*blockDim.x + threadIdx.x;
  if(idx >= H2*W2*64) return;
  int c  = idx & 63;
  int p  = idx >> 6;
  int xo = p & (W2-1);
  int yo = p >> 9;
  float sy = fminf(fmaxf((yo+0.5f)*0.5f-0.5f, 0.f), 255.f);
  float sx = fminf(fmaxf((xo+0.5f)*0.5f-0.5f, 0.f), 255.f);
  int y0 = (int)sy, x0 = (int)sx;
  int y1 = min(y0+1,255), x1 = min(x0+1,255);
  float fy = sy-(float)y0, fx = sx-(float)x0;
  const float* ch = x + (size_t)c*256*256;
  float v00=ch[y0*256+x0], v01=ch[y0*256+x1], v10=ch[y1*256+x0], v11=ch[y1*256+x1];
  out[idx] = (_Float16)(v00*(1.f-fy)*(1.f-fx)+v01*(1.f-fy)*fx+v10*fy*(1.f-fx)+v11*fy*fx);
}

// ---------------- implicit-GEMM conv3x3, weights TDM-staged in LDS ----------------
template<int CIN, int COUT, bool RELU>
__global__ void k_conv3x3(const _Float16* __restrict__ in, const _Float16* __restrict__ wpk,
                          const float* __restrict__ bias, _Float16* __restrict__ out){
  constexpr int NC = 9*CIN/32;
  __shared__ _Float16 wlds[NC*512];
  const int lane = threadIdx.x & 31;
  const int wv   = threadIdx.x >> 5;
  const int ct   = blockIdx.y;
  const int pt   = blockIdx.x*8 + wv;
  const int p0   = pt*16;
  const int y    = p0 >> 9;
  const int x0   = p0 & (W2-1);
  const int co0  = ct*16;
  const int m    = lane & 15;
  const int hi   = (lane & 16) ? 1 : 0;

  stage_weights(wlds, wpk + (size_t)ct*NC*512, NC*512);

  v8f acc = {};
  for(int tap=0; tap<9; ++tap){
    const int dy = tap/3 - 1, dx = tap%3 - 1;
    const int yy = y + dy;
    const int xx = x0 + m + dx;
    const bool inb = ((unsigned)yy < (unsigned)H2) && ((unsigned)xx < (unsigned)W2);
    const _Float16* src = in + ((size_t)yy*W2 + xx)*CIN + (hi?16:0);
#pragma unroll
    for(int cc=0; cc<CIN/32; ++cc){
      v16h a = load_a_frag(src + cc*32, inb);
      v16h b = ((const v16h*)wlds)[(size_t)(tap*(CIN/32)+cc)*32 + lane];
      acc = __builtin_amdgcn_wmma_f32_16x16x32_f16(false, a, false, b, (short)0, acc, false, false);
    }
  }
  const float bn = bias[co0 + m];
#pragma unroll
  for(int r=0;r<8;++r){
    float v = acc[r] + bn;
    if(RELU) v = fmaxf(v, 0.f);
    const int pid = p0 + r + hi*8;
    out[(size_t)pid*COUT + co0 + m] = (_Float16)v;
  }
}

// ---------------- token GEMM: Y = X @ W^T + b (then * oscale), weights TDM-staged ----------------
// OUTMODE: 0 = f16 token-major, 1 = f32 token-major, 2 = f16 channel-major (transposed, stride NPT)
template<int K, int N, int OUTMODE>
__global__ void k_gemm(const _Float16* __restrict__ X, const _Float16* __restrict__ wpk,
                       const float* __restrict__ bias, _Float16* __restrict__ Yh,
                       float* __restrict__ Yf, float oscale){
  constexpr int NC = K/32;
  __shared__ _Float16 wlds[NC*512];
  const int lane = threadIdx.x & 31;
  const int wv   = threadIdx.x >> 5;
  const int nt   = blockIdx.y;
  const int mt   = blockIdx.x*8 + wv;
  const int t0   = mt*16;
  const int n0   = nt*16;
  const int m    = lane & 15;
  const int hi   = (lane & 16) ? 1 : 0;

  stage_weights(wlds, wpk + (size_t)nt*NC*512, NC*512);

  v8f acc = {};
  const _Float16* xrow = X + (size_t)(t0+m)*K + (hi?16:0);
#pragma unroll
  for(int cc=0; cc<NC; ++cc){
    v16h a = load_a_frag(xrow + cc*32, true);
    v16h b = ((const v16h*)wlds)[(size_t)cc*32 + lane];
    acc = __builtin_amdgcn_wmma_f32_16x16x32_f16(false, a, false, b, (short)0, acc, false, false);
  }
  const float bn = bias[n0 + m];
#pragma unroll
  for(int r=0;r<8;++r){
    float v = (acc[r] + bn)*oscale;
    const int tok = t0 + r + hi*8;
    if(OUTMODE==1)      Yf[(size_t)tok*N + n0 + m] = v;
    else if(OUTMODE==2) Yh[(size_t)(n0+m)*NPT + tok] = (_Float16)v;
    else                Yh[(size_t)tok*N + n0 + m] = (_Float16)v;
  }
}

// ---------------- per-row L2 normalize (128 ch), one wave per row ----------------
__global__ void k_l2norm128(_Float16* __restrict__ d, int nrow){
  const int lane = threadIdx.x & 31;
  const int wid  = blockIdx.x*(blockDim.x>>5) + (threadIdx.x>>5);
  if(wid >= nrow) return;
  _Float16* row = d + (size_t)wid*128;
  float v[4]; float ss = 0.f;
#pragma unroll
  for(int i=0;i<4;++i){ v[i] = (float)row[lane+32*i]; ss += v[i]*v[i]; }
  ss = wred_sum32(ss);
  float inv = 1.f/fmaxf(sqrtf(ss), 1e-12f);
#pragma unroll
  for(int i=0;i<4;++i) row[lane+32*i] = (_Float16)(v[i]*inv);
}

// ---------------- pose MLP ----------------
__global__ void k_pose(const float* __restrict__ pose,
                       const float* __restrict__ w1, const float* __restrict__ b1,
                       const float* __restrict__ w2, const float* __restrict__ b2,
                       const float* __restrict__ pfw, const float* __restrict__ pfb,
                       float* __restrict__ pose_mod){
  __shared__ float h1[64];
  __shared__ float fp[64];
  const int t = threadIdx.x;            // 128 threads
  if(t < 64){
    float s = b1[t];
    for(int i=0;i<7;++i) s += w1[t*7+i]*pose[i];
    h1[t] = fmaxf(s, 0.f);
  }
  __syncthreads();
  if(t < 64){
    float s = b2[t];
    for(int i=0;i<64;++i) s += w2[t*64+i]*h1[i];
    fp[t] = s;
  }
  __syncthreads();
  float s = pfb[t];
  for(int i=0;i<64;++i) s += pfw[t*64+i]*fp[i];
  pose_mod[t] = s;
}

// ---------------- per-keypoint: patch mean, masked desc sample, var head, rel head ----------------
__global__ void k_point(const int* __restrict__ coords, const unsigned char* __restrict__ mask,
                        const _Float16* __restrict__ shmap, const _Float16* __restrict__ descmap,
                        const float* __restrict__ vw1, const float* __restrict__ vb1,
                        const float* __restrict__ vw2, const float* __restrict__ vb2,
                        const float* __restrict__ rw1, const float* __restrict__ rb1,
                        const float* __restrict__ rw2, const float* __restrict__ rb2,
                        _Float16* __restrict__ patch_out, _Float16* __restrict__ qin,
                        float* __restrict__ out_desc, float* __restrict__ out_rel,
                        float* __restrict__ out_var){
  __shared__ float red[128];
  const int n = blockIdx.x;
  const int t = threadIdx.x;           // 128 threads = channels
  const int y = coords[2*n+0];
  const int x = coords[2*n+1];

  float s = 0.f;
  for(int dy=-2;dy<=2;++dy)
    for(int dx=-2;dx<=2;++dx){
      int yy=y+dy, xx=x+dx;
      if((unsigned)yy<(unsigned)H2 && (unsigned)xx<(unsigned)W2)
        s += (float)shmap[((size_t)yy*W2+xx)*128 + t];
    }
  patch_out[(size_t)n*128+t] = (_Float16)(s*(1.f/25.f));

  float dv = 0.f;
  if((unsigned)y<(unsigned)H2 && (unsigned)x<(unsigned)W2)
    dv = (float)descmap[((size_t)y*W2+x)*128 + t];
  float dm = mask[(size_t)n*128+t] ? dv : 0.f;
  out_desc[(size_t)n*128+t] = dm;
  qin[(size_t)n*128+t] = (_Float16)dm;

  float contrib = 0.f;
  if(t < 64){
    float a = vb1[t];
    for(int tap=0;tap<9;++tap){
      int dy=tap/3-1, dx=tap%3-1;
      int yy=y+dy, xx=x+dx;
      if((unsigned)yy>=(unsigned)H2 || (unsigned)xx>=(unsigned)W2) continue;
      const _Float16* sp = shmap + ((size_t)yy*W2+xx)*128;
      const float* wp = vw1 + (size_t)t*128*9 + tap;
      for(int c=0;c<128;++c) a += (float)sp[c]*wp[(size_t)c*9];
    }
    contrib = fmaxf(a, 0.f)*vw2[t];
  }
  red[t] = contrib; __syncthreads();
  for(int st=64; st>0; st>>=1){ if(t<st) red[t]+=red[t+st]; __syncthreads(); }
  if(t==0){
    float xv = red[0] + vb2[0];
    out_var[n] = (xv > 20.f) ? xv : log1pf(__expf(xv));
  }
  __syncthreads();

  contrib = 0.f;
  if(t < 32){
    float a = rb1[t];
    for(int tap=0;tap<9;++tap){
      int dy=tap/3-1, dx=tap%3-1;
      int yy=y+dy, xx=x+dx;
      if((unsigned)yy>=(unsigned)H2 || (unsigned)xx>=(unsigned)W2) continue;
      const _Float16* sp = descmap + ((size_t)yy*W2+xx)*128;
      const float* wp = rw1 + (size_t)t*128*9 + tap;
      for(int c=0;c<128;++c) a += (float)sp[c]*wp[(size_t)c*9];
    }
    contrib = fmaxf(a, 0.f)*rw2[t];
  }
  red[t] = contrib; __syncthreads();
  for(int st=64; st>0; st>>=1){ if(t<st) red[t]+=red[t+st]; __syncthreads(); }
  if(t==0){
    float xv = red[0] + rb2[0];
    out_rel[n] = 1.f/(1.f+__expf(-xv));
  }
}

// ---------------- flash attention: one wave per (head, 16-row tile); 64 columns/iter ----------------
// qf: token-major, pre-scaled by 1/sqrt(dh). kf: token-major. vT: channel-major [128][4096].
// launch_bounds(128,1): only 1024 waves in this launch -> trade occupancy for zero spills.
__global__ void __launch_bounds__(128, 1)
k_attn(const _Float16* __restrict__ qf, const _Float16* __restrict__ kf,
       const _Float16* __restrict__ vT, _Float16* __restrict__ attn_out){
  __shared__ _Float16 pbuf[4][16*64];
  const int lane = threadIdx.x & 31;
  const int wv   = threadIdx.x >> 5;
  const int wid  = blockIdx.x*4 + wv;       // 4 heads * 256 row tiles
  const int h    = wid >> 8;
  const int it   = wid & 255;
  const int m    = lane & 15;
  const int hi   = (lane & 16) ? 1 : 0;

  // Q fragment: pure vector load (already scaled by q projection)
  v16h qa = load_a_frag(qf + (size_t)(it*16+m)*128 + h*32 + (hi?16:0), true);

  v8f o0 = {}, o1 = {};
  float mrow[8], lrow[8];
#pragma unroll
  for(int r=0;r<8;++r){ mrow[r] = -3.0e38f; lrow[r] = 0.f; }

  for(int jt=0; jt<NPT; jt+=64){
    // 4 K tiles as B operands (contiguous 32B per lane)
    v16h b0 = *((const v16h*)(kf + (size_t)(jt   +m)*128 + h*32) + hi);
    v16h b1 = *((const v16h*)(kf + (size_t)(jt+16+m)*128 + h*32) + hi);
    v16h b2 = *((const v16h*)(kf + (size_t)(jt+32+m)*128 + h*32) + hi);
    v16h b3 = *((const v16h*)(kf + (size_t)(jt+48+m)*128 + h*32) + hi);
    v8f s0={},s1={},s2={},s3={};
    s0 = __builtin_amdgcn_wmma_f32_16x16x32_f16(false, qa, false, b0, (short)0, s0, false, false);
    s1 = __builtin_amdgcn_wmma_f32_16x16x32_f16(false, qa, false, b1, (short)0, s1, false, false);
    s2 = __builtin_amdgcn_wmma_f32_16x16x32_f16(false, qa, false, b2, (short)0, s2, false, false);
    s3 = __builtin_amdgcn_wmma_f32_16x16x32_f16(false, qa, false, b3, (short)0, s3, false, false);

    // V B operands: issue early so the loads overlap the softmax VALU block below
    const _Float16* vr0 = vT + (size_t)(h*32 + m)*NPT      + jt + hi*16;
    const _Float16* vr1 = vT + (size_t)(h*32 + 16 + m)*NPT + jt + hi*16;
    v16h bv0a = *(const v16h*)(vr0);
    v16h bv0b = *(const v16h*)(vr0 + 32);
    v16h bv1a = *(const v16h*)(vr1);
    v16h bv1b = *(const v16h*)(vr1 + 32);

#pragma unroll
    for(int r=0;r<8;++r){
      float tm = wred_max16(fmaxf(fmaxf(s0[r],s1[r]), fmaxf(s2[r],s3[r])));
      float mn = fmaxf(mrow[r], tm);
      float sc = __expf(mrow[r]-mn);
      float p0 = __expf(s0[r]-mn);
      float p1 = __expf(s1[r]-mn);
      float p2 = __expf(s2[r]-mn);
      float p3 = __expf(s3[r]-mn);
      lrow[r] = lrow[r]*sc + wred_sum16(p0+p1+p2+p3);
      mrow[r] = mn;
      o0[r] *= sc; o1[r] *= sc;
      _Float16* pr = &pbuf[wv][(r + hi*8)*64];
      pr[m]      = (_Float16)p0;
      pr[16 + m] = (_Float16)p1;
      pr[32 + m] = (_Float16)p2;
      pr[48 + m] = (_Float16)p3;
    }
    // P (16x64) -> two A operands via vectorized LDS reloads
    v16h pa0, pa1;
    {
      union { u32x4 u[2]; v16h v; } au;
      const u32x4* pp = (const u32x4*)&pbuf[wv][m*64 + (hi?16:0)];
      au.u[0] = pp[0]; au.u[1] = pp[1]; pa0 = au.v;
      const u32x4* pq = (const u32x4*)&pbuf[wv][m*64 + 32 + (hi?16:0)];
      au.u[0] = pq[0]; au.u[1] = pq[1]; pa1 = au.v;
    }
    o0 = __builtin_amdgcn_wmma_f32_16x16x32_f16(false, pa0, false, bv0a, (short)0, o0, false, false);
    o0 = __builtin_amdgcn_wmma_f32_16x16x32_f16(false, pa1, false, bv0b, (short)0, o0, false, false);
    o1 = __builtin_amdgcn_wmma_f32_16x16x32_f16(false, pa0, false, bv1a, (short)0, o1, false, false);
    o1 = __builtin_amdgcn_wmma_f32_16x16x32_f16(false, pa1, false, bv1b, (short)0, o1, false, false);
  }
#pragma unroll
  for(int r=0;r<8;++r){
    float inv = 1.f/fmaxf(lrow[r], 1e-20f);
    const int tok = it*16 + r + hi*8;
    attn_out[(size_t)tok*128 + h*32 + m]      = (_Float16)(o0[r]*inv);
    attn_out[(size_t)tok*128 + h*32 + 16 + m] = (_Float16)(o1[r]*inv);
  }
}

// ---------------- fused = attn_out * (1 + pose_mod[c]) ----------------
__global__ void k_fuse_scale(const _Float16* __restrict__ attn, const float* __restrict__ pose_mod,
                             _Float16* __restrict__ outp){
  int idx = blockIdx.x*blockDim.x + threadIdx.x;
  if(idx >= NPT*128) return;
  int c = idx & 127;
  outp[idx] = (_Float16)((float)attn[idx]*(1.f+pose_mod[c]));
}

// ---------------- workspace layout (bytes) ----------------
#define OFF_XUP     ((size_t)0)
#define OFF_ACT1    ((size_t)33554432)
#define OFF_SHARED  ((size_t)67108864)
#define OFF_DESC    ((size_t)134217728)
#define OFF_PATCH   ((size_t)201326592)
#define OFF_QIN     ((size_t)202375168)
#define OFF_QF      ((size_t)203423744)
#define OFF_KF      ((size_t)204472320)
#define OFF_VT      ((size_t)205520896)
#define OFF_ATTN    ((size_t)206569472)
#define OFF_FUSED   ((size_t)207618048)
#define OFF_LATENT  ((size_t)208666624)
#define OFF_POSE    ((size_t)209715200)
#define OFF_WPK_BB1 ((size_t)209716224)
#define OFF_WPK_BB2 ((size_t)209789952)
#define OFF_WPK_DSC ((size_t)209937408)
#define OFF_WPK_Q   ((size_t)210232320)
#define OFF_WPK_K   ((size_t)210265088)
#define OFF_WPK_V   ((size_t)210297856)
#define OFF_WPK_OF  ((size_t)210330624)
#define OFF_WPK_DEC ((size_t)210363392)

extern "C" void kernel_launch(void* const* d_in, const int* in_sizes, int n_in,
                              void* d_out, int out_size, void* d_ws, size_t ws_size,
                              hipStream_t stream) {
  (void)in_sizes; (void)n_in; (void)out_size; (void)ws_size;
  const float* xfeat = (const float*)d_in[0];
  const float* pose  = (const float*)d_in[1];
  const int*   coords= (const int*)d_in[2];
  const unsigned char* dmask = (const unsigned char*)d_in[3];
  const float* bb_w1 = (const float*)d_in[4];  const float* bb_b1 = (const float*)d_in[5];
  const float* bb_w2 = (const float*)d_in[6];  const float* bb_b2 = (const float*)d_in[7];
  const float* var_w1= (const float*)d_in[8];  const float* var_b1= (const float*)d_in[9];
  const float* var_w2= (const float*)d_in[10]; const float* var_b2= (const float*)d_in[11];
  const float* desc_w= (const float*)d_in[12]; const float* desc_b= (const float*)d_in[13];
  const float* rel_w1= (const float*)d_in[14]; const float* rel_b1= (const float*)d_in[15];
  const float* rel_w2= (const float*)d_in[16]; const float* rel_b2= (const float*)d_in[17];
  const float* pe_w1 = (const float*)d_in[18]; const float* pe_b1 = (const float*)d_in[19];
  const float* pe_w2 = (const float*)d_in[20]; const float* pe_b2 = (const float*)d_in[21];
  const float* q_w   = (const float*)d_in[22]; const float* q_b   = (const float*)d_in[23];
  const float* k_w   = (const float*)d_in[24]; const float* k_b   = (const float*)d_in[25];
  const float* v_w   = (const float*)d_in[26]; const float* v_b   = (const float*)d_in[27];
  const float* pf_w  = (const float*)d_in[28]; const float* pf_b  = (const float*)d_in[29];
  const float* of_w  = (const float*)d_in[30]; const float* of_b  = (const float*)d_in[31];
  const float* dec_w = (const float*)d_in[32]; const float* dec_b = (const float*)d_in[33];

  char* ws = (char*)d_ws;
  _Float16* xup    = (_Float16*)(ws + OFF_XUP);
  _Float16* act1   = (_Float16*)(ws + OFF_ACT1);
  _Float16* shmap  = (_Float16*)(ws + OFF_SHARED);
  _Float16* descm  = (_Float16*)(ws + OFF_DESC);
  _Float16* patchf = (_Float16*)(ws + OFF_PATCH);
  _Float16* qin    = (_Float16*)(ws + OFF_QIN);
  _Float16* qf     = (_Float16*)(ws + OFF_QF);
  _Float16* kfb    = (_Float16*)(ws + OFF_KF);
  _Float16* vT     = (_Float16*)(ws + OFF_VT);
  _Float16* attn   = (_Float16*)(ws + OFF_ATTN);
  _Float16* fusedi = (_Float16*)(ws + OFF_FUSED);
  _Float16* latent = (_Float16*)(ws + OFF_LATENT);
  float*    posem  = (float*)(ws + OFF_POSE);
  _Float16* wpk_bb1= (_Float16*)(ws + OFF_WPK_BB1);
  _Float16* wpk_bb2= (_Float16*)(ws + OFF_WPK_BB2);
  _Float16* wpk_dsc= (_Float16*)(ws + OFF_WPK_DSC);
  _Float16* wpk_q  = (_Float16*)(ws + OFF_WPK_Q);
  _Float16* wpk_k  = (_Float16*)(ws + OFF_WPK_K);
  _Float16* wpk_v  = (_Float16*)(ws + OFF_WPK_V);
  _Float16* wpk_of = (_Float16*)(ws + OFF_WPK_OF);
  _Float16* wpk_dec= (_Float16*)(ws + OFF_WPK_DEC);

  float* out      = (float*)d_out;
  float* out_pred = out;                 // 4096*64
  float* out_desc = out + 262144;        // 4096*128
  float* out_rel  = out + 786432;        // 4096
  float* out_var  = out + 790528;        // 4096

  const float qscale = 0.17677669529663687f;   // 1/sqrt(32)

  // 0) repack all weights into f16 WMMA B-operand layout
  k_repack<64,64,9>   <<<144, 256, 0, stream>>>(bb_w1,  wpk_bb1);
  k_repack<64,128,9>  <<<288, 256, 0, stream>>>(bb_w2,  wpk_bb2);
  k_repack<128,128,9> <<<576, 256, 0, stream>>>(desc_w, wpk_dsc);
  k_repack<128,128,1> <<< 64, 256, 0, stream>>>(q_w,    wpk_q);
  k_repack<128,128,1> <<< 64, 256, 0, stream>>>(k_w,    wpk_k);
  k_repack<128,128,1> <<< 64, 256, 0, stream>>>(v_w,    wpk_v);
  k_repack<128,128,1> <<< 64, 256, 0, stream>>>(of_w,   wpk_of);
  k_repack<128,64,1>  <<< 32, 256, 0, stream>>>(dec_w,  wpk_dec);

  // 1) upsample to 512x512 HWC f16
  k_upsample<<<(H2*W2*64 + 255)/256, 256, 0, stream>>>(xfeat, xup);
  // 2) backbone convs (WMMA implicit GEMM, TDM weight staging)
  k_conv3x3<64,64,true >  <<<dim3(2048,4), 256, 0, stream>>>(xup,  wpk_bb1, bb_b1, act1);
  k_conv3x3<64,128,true>  <<<dim3(2048,8), 256, 0, stream>>>(act1, wpk_bb2, bb_b2, shmap);
  // 3) descriptor conv + per-pixel L2 norm
  k_conv3x3<128,128,false><<<dim3(2048,8), 256, 0, stream>>>(shmap, wpk_dsc, desc_b, descm);
  k_l2norm128<<<32768, 256, 0, stream>>>(descm, H2*W2);
  // 4) pose MLP
  k_pose<<<1, 128, 0, stream>>>(pose, pe_w1, pe_b1, pe_w2, pe_b2, pf_w, pf_b, posem);
  // 5) per-keypoint sampling + var/rel heads + patch means
  k_point<<<NPT, 128, 0, stream>>>(coords, dmask, shmap, descm,
                                   var_w1, var_b1, var_w2, var_b2,
                                   rel_w1, rel_b1, rel_w2, rel_b2,
                                   patchf, qin, out_desc, out_rel, out_var);
  // 6) Q/K/V projections; Q pre-scaled by 1/sqrt(dh); V stored transposed (channel-major)
  k_gemm<128,128,0><<<dim3(32,8), 256, 0, stream>>>(qin,    wpk_q, q_b, qf,  nullptr, qscale);
  k_gemm<128,128,0><<<dim3(32,8), 256, 0, stream>>>(patchf, wpk_k, k_b, kfb, nullptr, 1.0f);
  k_gemm<128,128,2><<<dim3(32,8), 256, 0, stream>>>(patchf, wpk_v, v_b, vT,  nullptr, 1.0f);
  // 7) flash attention (WMMA), 4 heads x 256 row tiles, 64 cols/iter
  k_attn<<<256, 128, 0, stream>>>(qf, kfb, vT, attn);
  // 8) pose modulation, output projection, L2 norm, decoder
  k_fuse_scale<<<(NPT*128 + 255)/256, 256, 0, stream>>>(attn, posem, fusedi);
  k_gemm<128,128,0><<<dim3(32,8), 256, 0, stream>>>(fusedi, wpk_of, of_b, latent, nullptr, 1.0f);
  k_l2norm128<<<512, 256, 0, stream>>>(latent, NPT);
  k_gemm<128,64,1> <<<dim3(32,4), 256, 0, stream>>>(latent, wpk_dec, dec_b, nullptr, out_pred, 1.0f);
}
